// ProxySpatialAttentionModule_77197742178862
// MI455X (gfx1250) — matrix operand order repaired and checked by reading, not compiled, not run on hardware
//
#include <hip/hip_runtime.h>
#include <hip/hip_bf16.h>

// ProxySpatialAttention for MI455X (gfx1250, wave32, WMMA bf16 16x16x32).
// B=4, C=256, H=W=56 -> G=8 heads, hd=32, L=3136. BG = 32 (b,g) problems.
//
//   k1: grouped 1x1-conv QKV projection (f32 math, bf16 pair-packed outputs)
//   k2: per-row softmax stats, TWO-PHASE (max sweep, then sum-exp sweep):
//       1 exp per energy element instead of 2 (v_exp_f32 is 1/4-rate TRANS,
//       WMMA re-sweep of L2-resident K is cheaper than the extra exps)
//   k3: out = V * softmax(Q^T K) + x, E recomputed in WMMA, P staged in LDS
//
// Q/K are stored channel-pair interleaved so every WMMA fragment pair is one
// aligned b32 load; V is [d][l] (its fragment pairs are j-adjacent). The
// 3136x3136 energy matrix (1.26 GB across batch*heads) is never materialized;
// Q/K/V (~600KB per (b,g)) stay L2-resident (192MB L2).

#define HEADS 8
#define HD    32
#define HDP   16              // HD/2 channel pairs
#define LSP   3136            // 56*56
#define LSPH  1568            // LSP/2
#define BGN   32              // B * HEADS
#define KT16  196             // LSP / 16
#define JCH32 98              // LSP / 32

typedef __attribute__((ext_vector_type(16))) __bf16 v16bf;
typedef __attribute__((ext_vector_type(8)))  float  v8f;

union BF16x16 { v16bf v; __bf16 e[16]; unsigned int u[8]; };
union F32x8   { v8f   v; float  e[8];  };

static __device__ __forceinline__ v8f wmma_bf16(const v16bf& a, const v16bf& b, const v8f& c) {
    return __builtin_amdgcn_wmma_f32_16x16x32_bf16(
        /*neg_a=*/false, a, /*neg_b=*/false, b,
        /*c_mod=*/(short)0, c, /*reuse_a=*/false, /*reuse_b=*/false);
}

static __device__ __forceinline__ unsigned int pack_bf16(float a, float b) {
    const unsigned short lo = __builtin_bit_cast(unsigned short, (__bf16)a);
    const unsigned short hi = __builtin_bit_cast(unsigned short, (__bf16)b);
    return (unsigned int)lo | ((unsigned int)hi << 16);
}

// ---------------------------------------------------------------------------
// Kernel 1: per-head 1x1 conv projections. One thread = one spatial location,
// all 32 channels, all three of Q/K/V (3*32*32 FMAs). Weights are block-
// uniform -> scalar loads. Q/K written as channel-pair uints (coalesced b32);
// V written bf16 [bg][d][l].
// ---------------------------------------------------------------------------
__global__ __launch_bounds__(256) void qkv_proj_kernel(
    const float* __restrict__ x,
    const float* __restrict__ Wq, const float* __restrict__ bq,
    const float* __restrict__ Wk, const float* __restrict__ bk,
    const float* __restrict__ Wv, const float* __restrict__ bv,
    unsigned int* __restrict__ q, unsigned int* __restrict__ k,
    __bf16* __restrict__ v)
{
    const int bg = blockIdx.x;
    const int l  = blockIdx.y * 256 + threadIdx.x;
    if (l >= LSP) return;
    const int g = bg & (HEADS - 1);

    const float* xp = x + (size_t)bg * HD * LSP + l;
    float xi[HD];
    #pragma unroll
    for (int i = 0; i < HD; ++i) xi[i] = xp[(size_t)i * LSP];

    const float* wq = Wq + g * HD * HD;
    const float* wk = Wk + g * HD * HD;
    const float* wv = Wv + g * HD * HD;

    for (int o = 0; o < HD; o += 2) {
        float aq0 = bq[g * HD + o],     ak0 = bk[g * HD + o],     av0 = bv[g * HD + o];
        float aq1 = bq[g * HD + o + 1], ak1 = bk[g * HD + o + 1], av1 = bv[g * HD + o + 1];
        #pragma unroll
        for (int i = 0; i < HD; ++i) {
            aq0 = fmaf(wq[o * HD + i],       xi[i], aq0);
            aq1 = fmaf(wq[(o + 1) * HD + i], xi[i], aq1);
            ak0 = fmaf(wk[o * HD + i],       xi[i], ak0);
            ak1 = fmaf(wk[(o + 1) * HD + i], xi[i], ak1);
            av0 = fmaf(wv[o * HD + i],       xi[i], av0);
            av1 = fmaf(wv[(o + 1) * HD + i], xi[i], av1);
        }
        const size_t pidx = (size_t)(bg * HDP + (o >> 1)) * LSP + l;
        q[pidx] = pack_bf16(aq0, aq1);
        k[pidx] = pack_bf16(ak0, ak1);
        v[(size_t)(bg * HD + o) * LSP + l]     = (__bf16)av0;
        v[(size_t)(bg * HD + o + 1) * LSP + l] = (__bf16)av1;
    }
}

// Fragment loaders (channel-pair uint layout, element pair (d, d+1) at [d>>1][l]).
// A operand (M=16 j/d rows x K=32 channels): lane = M, vgpr r holds
// K = 2(r&3) + 16(r>=4) + 8h -- always an even channel -> one u32.
static __device__ __forceinline__ void load_A_chan(
    BF16x16& a, const unsigned int* __restrict__ base, int row, int h)
{
    #pragma unroll
    for (int r = 0; r < 8; ++r) {
        const int kk0 = ((r & 3) << 1) + ((r >> 2) << 4) + (h << 3);
        a.u[r] = base[(size_t)(kk0 >> 1) * LSP + row];
    }
}
// B operand (K=32 channels x N=16 cols): lane = N, vgpr r holds K = 2(r&3) + 16(r>=4) + 8h.
static __device__ __forceinline__ void load_B_chan(
    BF16x16& b, const unsigned int* __restrict__ base, int col, int h)
{
    #pragma unroll
    for (int r = 0; r < 8; ++r) {
        const int d0 = ((r & 3) << 1) + ((r >> 2) << 4) + (h << 3);
        b.u[r] = base[(size_t)(d0 >> 1) * LSP + col];
    }
}

// ---------------------------------------------------------------------------
// Kernel 2: row softmax stats, two-phase. One wave owns a 16-row j-tile.
// Phase A: sweep 196 k-tiles with WMMA, track per-lane partial row max only.
// Butterfly-reduce max across the 16-lane accumulator halves (all lanes end
// up holding their rows' maxima). Phase B: sweep again, ps += exp(e - m).
// Butterfly-reduce the sums. stats[j] = (m_j, 1/Z_j).
// ---------------------------------------------------------------------------
__global__ __launch_bounds__(256) void row_stats_kernel(
    const unsigned int* __restrict__ q, const unsigned int* __restrict__ k,
    float2* __restrict__ stats)
{
    const int bg   = blockIdx.x;
    const int wave = threadIdx.x >> 5;
    const int jt16 = blockIdx.y * 8 + wave;
    if (jt16 >= KT16) return;
    const int j0   = jt16 * 16;
    const int lane = threadIdx.x & 31;
    const int n    = lane & 15;
    const int h    = lane >> 4;

    const unsigned int* qb = q + (size_t)bg * HDP * LSP;
    const unsigned int* kb = k + (size_t)bg * HDP * LSP;

    BF16x16 qa;                       // Q^T tile, loop-invariant
    load_A_chan(qa, qb, j0 + n, h);

    // ---- Phase A: row maxima (no transcendentals in the loop) ----
    float pm[8];
    #pragma unroll
    for (int r = 0; r < 8; ++r) pm[r] = -1e30f;

    for (int kt = 0; kt < KT16; ++kt) {
        BF16x16 kf;
        load_B_chan(kf, kb, kt * 16 + n, h);
        v8f cz = {};
        F32x8 e; e.v = wmma_bf16(qa.v, kf.v, cz);
        #pragma unroll
        for (int r = 0; r < 8; ++r) pm[r] = fmaxf(pm[r], e.e[r]);
    }
    #pragma unroll
    for (int r = 0; r < 8; ++r) {
        #pragma unroll
        for (int d = 1; d < 16; d <<= 1)
            pm[r] = fmaxf(pm[r], __shfl_xor(pm[r], d, 32));
    }

    // ---- Phase B: sum of exp(e - m): exactly 1 exp per energy element ----
    float ps[8];
    #pragma unroll
    for (int r = 0; r < 8; ++r) ps[r] = 0.0f;

    for (int kt = 0; kt < KT16; ++kt) {
        BF16x16 kf;
        load_B_chan(kf, kb, kt * 16 + n, h);
        v8f cz = {};
        F32x8 e; e.v = wmma_bf16(qa.v, kf.v, cz);
        #pragma unroll
        for (int r = 0; r < 8; ++r) ps[r] += __expf(e.e[r] - pm[r]);
    }
    #pragma unroll
    for (int r = 0; r < 8; ++r) {
        #pragma unroll
        for (int d = 1; d < 16; d <<= 1)
            ps[r] += __shfl_xor(ps[r], d, 32);
        if (n == 0) {
            const int jrow = j0 + (h << 3) + r;
            stats[(size_t)bg * LSP + jrow] = make_float2(pm[r], 1.0f / ps[r]);
        }
    }
}

// ---------------------------------------------------------------------------
// Kernel 3: out[:, k-tile] = sum_j V[:, j] * exp(E[j,k]-m_j)/Z_j  + residual.
// One wave owns a 16-col k-tile (K-operand loop-invariant). Per 32-j chunk:
//   2x WMMA  -> E tiles (f32 accum layout)
//   exp+scale -> bf16 P pairs packed, 4x ds_store_b32 per tile into LDS
//   2x WMMA  -> acc(32x16) += V_tile * P   (B operand: 8x ds_load_b32)
// ---------------------------------------------------------------------------
__global__ __launch_bounds__(256) void attn_out_kernel(
    const unsigned int* __restrict__ q, const unsigned int* __restrict__ k,
    const __bf16* __restrict__ v, const float2* __restrict__ stats,
    const float* __restrict__ x, float* __restrict__ out)
{
    __shared__ __bf16 pl[8][16][32];    // per-wave 16(n) x 32(j_local) P tile

    const int bg   = blockIdx.x;
    const int wave = threadIdx.x >> 5;
    const int kt   = blockIdx.y * 8 + wave;
    if (kt >= KT16) return;
    const int k0   = kt * 16;
    const int lane = threadIdx.x & 31;
    const int n    = lane & 15;
    const int h    = lane >> 4;

    const unsigned int* qb = q + (size_t)bg * HDP * LSP;
    const unsigned int* kb = k + (size_t)bg * HDP * LSP;
    const unsigned int* vu = (const unsigned int*)v + (size_t)bg * HD * LSPH;
    const float2*       st = stats + (size_t)bg * LSP;

    BF16x16 kf;                        // K tile, loop-invariant
    load_B_chan(kf, kb, k0 + n, h);

    F32x8 acc0, acc1;
    #pragma unroll
    for (int r = 0; r < 8; ++r) { acc0.e[r] = 0.0f; acc1.e[r] = 0.0f; }

    for (int jc = 0; jc < JCH32; ++jc) {
        const int j0 = jc * 32;

        // --- E tiles -> P (bf16 pairs) in LDS ---
        #pragma unroll
        for (int t = 0; t < 2; ++t) {
            const int jt = j0 + t * 16;
            BF16x16 qa;
            load_A_chan(qa, qb, jt + n, h);
            v8f cz = {};
            F32x8 e; e.v = wmma_bf16(qa.v, kf.v, cz);
            // rows jl = 8h + 2i, 8h + 2i + 1 are contiguous -> b32 stores
            #pragma unroll
            for (int i = 0; i < 4; ++i) {
                const int jl = (h << 3) + 2 * i;
                const float2 ms0 = st[jt + jl];
                const float2 ms1 = st[jt + jl + 1];
                const float p0 = __expf(e.e[2 * i]     - ms0.x) * ms0.y;
                const float p1 = __expf(e.e[2 * i + 1] - ms1.x) * ms1.y;
                *(unsigned int*)&pl[wave][n][t * 16 + jl] = pack_bf16(p0, p1);
            }
        }
        // Same-wave LDS RAW: drain DS counter before re-reading the tile.
        asm volatile("s_wait_dscnt 0x0" ::: "memory");

        // B operand: P (32 j x 16 k) re-read in WMMA B layout (b32 loads).
        BF16x16 pb;
        #pragma unroll
        for (int r = 0; r < 8; ++r) {
            const int kk0 = ((r & 3) << 1) + ((r >> 2) << 4) + (h << 3);
            pb.u[r] = *(const unsigned int*)&pl[wave][n][kk0];
        }

        // --- acc += V * P (two 16-row d-tiles); V pairs are j-adjacent ---
        #pragma unroll
        for (int t = 0; t < 2; ++t) {
            const int d = n + 16 * t;      // A operand: lane = M row (d)
            BF16x16 va;
            #pragma unroll
            for (int r = 0; r < 8; ++r) {
                const int kk0 = ((r & 3) << 1) + ((r >> 2) << 4) + (h << 3);
                va.u[r] = vu[(size_t)d * LSPH + ((j0 + kk0) >> 1)];
            }
            if (t == 0) acc0.v = wmma_bf16(va.v, pb.v, acc0.v);
            else        acc1.v = wmma_bf16(va.v, pb.v, acc1.v);
        }
    }

    // Epilogue: residual add + store (f32).
    #pragma unroll
    for (int t = 0; t < 2; ++t) {
        #pragma unroll
        for (int r = 0; r < 8; ++r) {
            const int d = t * 16 + (h << 3) + r;
            const size_t idx = (size_t)(bg * HD + d) * LSP + k0 + n;
            const float a = (t == 0) ? acc0.e[r] : acc1.e[r];
            out[idx] = a + x[idx];
        }
    }
}

// ---------------------------------------------------------------------------
extern "C" void kernel_launch(void* const* d_in, const int* in_sizes, int n_in,
                              void* d_out, int out_size, void* d_ws, size_t ws_size,
                              hipStream_t stream)
{
    (void)in_sizes; (void)n_in; (void)out_size; (void)ws_size;
    const float* x  = (const float*)d_in[0];
    const float* Wq = (const float*)d_in[1];
    const float* bq = (const float*)d_in[2];
    const float* Wk = (const float*)d_in[3];
    const float* bk = (const float*)d_in[4];
    const float* Wv = (const float*)d_in[5];
    const float* bv = (const float*)d_in[6];
    float* out = (float*)d_out;

    // Workspace: Q,K (uint pair-packed) + V (bf16) + stats float2  (~20 MB)
    const size_t qkv_bytes = (size_t)BGN * HD * LSP * sizeof(__bf16); // 6,422,528
    char* ws = (char*)d_ws;
    unsigned int* qw = (unsigned int*)(ws);
    unsigned int* kw = (unsigned int*)(ws + qkv_bytes);
    __bf16*       vw = (__bf16*)(ws + 2 * qkv_bytes);
    float2*       st = (float2*)(ws + 3 * qkv_bytes);

    qkv_proj_kernel<<<dim3(BGN, (LSP + 255) / 256), 256, 0, stream>>>(
        x, Wq, bq, Wk, bk, Wv, bv, qw, kw, vw);

    row_stats_kernel<<<dim3(BGN, (KT16 + 7) / 8), 256, 0, stream>>>(qw, kw, st);

    attn_out_kernel<<<dim3(BGN, (KT16 + 7) / 8), 256, 0, stream>>>(
        qw, kw, vw, st, x, out);
}
